// MTFMTextModel_23905787969915
// MI455X (gfx1250) — matrix operand order repaired and checked by reading, not compiled, not run on hardware
//
#include <hip/hip_runtime.h>
#include <cstdint>
#include <cstddef>

// ---------------------------------------------------------------------------
// MI455X / gfx1250, wave32, WMMA f16 -> f32 (16x16x32)
// ---------------------------------------------------------------------------
typedef __attribute__((ext_vector_type(16))) _Float16 v16h;
typedef __attribute__((ext_vector_type(8)))  float    v8f;

union Frag16 { v16h v; float4 f4[2]; };

#define WMMA_F16F32(a, b, c) \
  __builtin_amdgcn_wmma_f32_16x16x32_f16(false, (a), false, (b), (short)0, (c), false, false)

#define B_    256
#define S_    512
#define EP_   128   // E=100 padded to 128
#define KCH_  256   // conv output channels
#define A_    1000
#define APAD_ 1008  // 63 N-tiles of 16
#define NT_   63

// ---------------------------------------------------------------------------
// zero words
// ---------------------------------------------------------------------------
__global__ void mtfm_zero_words(uint32_t* __restrict__ p, long n) {
  long i = (long)blockIdx.x * blockDim.x + threadIdx.x;
  if (i < n) p[i] = 0u;
}

// ---------------------------------------------------------------------------
// embedding gather: Xemb[b,s,0:128] = f16(emb_table[ids[b,s], 0:100]), pad 0
// ---------------------------------------------------------------------------
__global__ void mtfm_embed_gather(const int* __restrict__ ids,
                                  const float* __restrict__ table,
                                  _Float16* __restrict__ X) {
  const int s = blockIdx.x, b = blockIdx.y, e = threadIdx.x;   // 128 threads
  const int id = ids[b * S_ + s];
  float v = (e < 100) ? table[(size_t)id * 100 + e] : 0.0f;
  X[((size_t)b * S_ + s) * EP_ + e] = (_Float16)v;
}

// ---------------------------------------------------------------------------
// weight packing
// ---------------------------------------------------------------------------
// conv_w (K, E=100, h) f32 -> Wp (K, h*128) f16, Wp[k, j*128+e] = w[k,e,j]
__global__ void mtfm_pack_conv(const float* __restrict__ src,
                               _Float16* __restrict__ dst, int h) {
  const int k = blockIdx.x, e = threadIdx.x;                   // 256 x 128
  for (int j = 0; j < h; ++j) {
    float v = (e < 100) ? src[((size_t)k * 100 + e) * h + j] : 0.0f;
    dst[(size_t)k * (h * EP_) + j * EP_ + e] = (_Float16)v;
  }
}

// (rowsReal, colsReal) f32 -> (APAD_, colsPad) f16, zero padded
__global__ void mtfm_pack_pad(const float* __restrict__ src,
                              _Float16* __restrict__ dst,
                              int rowsReal, int colsReal, int colsPad) {
  const int a = blockIdx.x;                                    // APAD_
  for (int j = threadIdx.x; j < colsPad; j += blockDim.x) {
    float v = (a < rowsReal && j < colsReal) ? src[(size_t)a * colsReal + j] : 0.0f;
    dst[(size_t)a * colsPad + j] = (_Float16)v;
  }
}

// (1000, 2000) f32 -> (APAD_, 2048) f16; halves placed at col 0 and col 1024
__global__ void mtfm_pack_split(const float* __restrict__ src,
                                _Float16* __restrict__ dst) {
  const int a = blockIdx.x;                                    // APAD_
  for (int j = threadIdx.x; j < 2048; j += blockDim.x) {
    float v = 0.0f;
    if (a < A_) {
      if (j < A_)                       v = src[(size_t)a * 2000 + j];
      else if (j >= 1024 && j < 2024)   v = src[(size_t)a * 2000 + 1000 + (j - 1024)];
    }
    dst[(size_t)a * 2048 + j] = (_Float16)v;
  }
}

// ---------------------------------------------------------------------------
// conv-as-GEMM via WMMA + relu + max-pool (atomic max, values >= 0)
// block = 256 thr (8 waves); wave w: t-tile = 16*(w>>2), k-tile = 64*(w&3)
// grid = (ceil(Lh/32), B)
// ---------------------------------------------------------------------------
__global__ __launch_bounds__(256) void mtfm_conv_wmma(
    const _Float16* __restrict__ X,    // (B, S, 128)
    const _Float16* __restrict__ Wp,   // (256, h*128)
    const float*    __restrict__ bias, // (256)
    float*          __restrict__ pool, // (B, 1024)
    int h, int Lh, int segOff) {
  const int KL   = h * EP_;
  const int b    = blockIdx.y;
  const int lane = threadIdx.x & 31;
  const int wave = threadIdx.x >> 5;
  const int half = lane >> 4;
  const int l15  = lane & 15;
  const int tbase = blockIdx.x * 32 + (wave >> 2) * 16;
  const int kbase = (wave & 3) * 64;

  int tA = tbase + l15;                 // A-matrix row (window position)
  if (tA > Lh - 1) tA = Lh - 1;         // clamp tail (stores are guarded)
  const _Float16* arow = X + ((size_t)b * S_ + tA) * EP_;

  v8f acc[4] = {};
  for (int kk = 0; kk < KL; kk += 32) {
    Frag16 af;
    af.f4[0] = *(const float4*)(arow + kk + half * 8);
    af.f4[1] = *(const float4*)(arow + kk + 16 + half * 8);
    __builtin_prefetch(arow + kk + 32, 0, 1);
#pragma unroll
    for (int j = 0; j < 4; ++j) {
      const _Float16* wrow = Wp + (size_t)(kbase + j * 16 + l15) * KL + kk + half * 8;
      Frag16 bf;
      bf.f4[0] = *(const float4*)(wrow);
      bf.f4[1] = *(const float4*)(wrow + 16);
      acc[j] = WMMA_F16F32(af.v, bf.v, acc[j]);
    }
  }

#pragma unroll
  for (int j = 0; j < 4; ++j) {
    const int k = kbase + j * 16 + l15;
    const float bs = bias[k];
    int* dst = (int*)(pool + (size_t)b * 1024 + segOff + k);
#pragma unroll
    for (int r = 0; r < 8; ++r) {
      const int t = tbase + r + half * 8;
      if (t < Lh) {
        float v = fmaxf(acc[j][r] + bs, 0.0f);
        atomicMax(dst, __float_as_int(v));   // valid: v >= 0, pool init 0
      }
    }
  }
}

// ---------------------------------------------------------------------------
// generic WMMA GEMM: out(256, Nreal) = A(256,Kdim) @ W(APAD_,Kdim)^T + bias
// act: 0 = none, 1 = tanh.  Writes f32 (ld=Nreal) and/or f16 (ld=outLd,+colOff)
// grid = (16, 8), block = 256 (8 waves, one 16x16 tile each)
// ---------------------------------------------------------------------------
__global__ __launch_bounds__(256) void mtfm_gemm_wmma(
    const _Float16* __restrict__ A,
    const _Float16* __restrict__ W,
    const float*    __restrict__ bias,
    int Kdim, int Nreal, int nTiles, int act,
    float* __restrict__ outF32,
    _Float16* __restrict__ outF16, int outLd, int colOff) {
  const int lane = threadIdx.x & 31;
  const int wave = threadIdx.x >> 5;
  const int half = lane >> 4;
  const int l15  = lane & 15;
  const int nTile = blockIdx.y * 8 + wave;
  if (nTile >= nTiles) return;
  const int mbase = blockIdx.x * 16;

  const _Float16* arow = A + (size_t)(mbase + l15) * Kdim;
  const _Float16* wrow = W + (size_t)(nTile * 16 + l15) * Kdim;

  v8f acc = {};
  for (int kk = 0; kk < Kdim; kk += 32) {
    Frag16 af, bf;
    af.f4[0] = *(const float4*)(arow + kk + half * 8);
    af.f4[1] = *(const float4*)(arow + kk + 16 + half * 8);
    bf.f4[0] = *(const float4*)(wrow + kk + half * 8);
    bf.f4[1] = *(const float4*)(wrow + kk + 16 + half * 8);
    __builtin_prefetch(wrow + kk + 32, 0, 1);
    acc = WMMA_F16F32(af.v, bf.v, acc);
  }

  const int n = nTile * 16 + l15;
  const float bs = (n < Nreal) ? bias[n] : 0.0f;
#pragma unroll
  for (int r = 0; r < 8; ++r) {
    const int m = mbase + r + half * 8;
    float v = acc[r] + bs;
    if (act == 1) v = tanhf(v);
    if (n < Nreal) {
      if (outF32) outF32[(size_t)m * Nreal + n] = v;
      if (outF16) outF16[(size_t)m * outLd + colOff + n] = (_Float16)v;
    }
  }
}

// ---------------------------------------------------------------------------
// e -> f16 copy
// ---------------------------------------------------------------------------
__global__ void mtfm_f32_to_f16(const float* __restrict__ s,
                                _Float16* __restrict__ d, int n) {
  int i = blockIdx.x * blockDim.x + threadIdx.x;
  if (i < n) d[i] = (_Float16)s[i];
}

// ---------------------------------------------------------------------------
// u_t (256, 8) = e (256,1024) @ fic_fc_w(8,1024)^T + b
// ---------------------------------------------------------------------------
__global__ void mtfm_ut(const float* __restrict__ e,
                        const float* __restrict__ w,
                        const float* __restrict__ b,
                        float* __restrict__ ut) {
  const int bb = blockIdx.x, f = threadIdx.x;    // 256 x 32
  if (f < 8) {
    float acc = b[f];
    const float* er = e + (size_t)bb * 1024;
    const float* wr = w + (size_t)f * 1024;
    for (int j = 0; j < 1024; ++j) acc += er[j] * wr[j];
    ut[bb * 8 + f] = acc;
  }
}

// ---------------------------------------------------------------------------
// head: u_mm = u_t @ api_emb ; u_mlp = tanh(sum_f w2[f]*(p[b,f]+q[a,f]+b1[f])+b2)
// writes fic_in f16 (256, 2048): [u_mm | pad | u_mlp | pad]
// ---------------------------------------------------------------------------
__global__ __launch_bounds__(256) void mtfm_head(
    const float* __restrict__ ut,    // (256,8)
    const float* __restrict__ api,   // (8,1000)
    const float* __restrict__ w1,    // (8,16)
    const float* __restrict__ b1,    // (8)
    const float* __restrict__ w2,    // (8)
    const float* __restrict__ b2,    // (1)
    _Float16* __restrict__ ficin) {  // (256,2048)
  __shared__ float sut[8], sp[8], sw1[128];
  const int b = blockIdx.x, tid = threadIdx.x;
  if (tid < 8)   sut[tid] = ut[b * 8 + tid];
  if (tid < 128) sw1[tid] = w1[tid];
  __syncthreads();
  if (tid < 8) {
    float acc = 0.0f;
    for (int f2 = 0; f2 < 8; ++f2) acc += sut[f2] * sw1[tid * 16 + f2];
    sp[tid] = acc;
  }
  __syncthreads();
  const float bb2 = b2[0];
  for (int a = tid; a < A_; a += 256) {
    float ae[8], umm = 0.0f;
#pragma unroll
    for (int f = 0; f < 8; ++f) { ae[f] = api[f * A_ + a]; umm += sut[f] * ae[f]; }
    float acc = bb2;
#pragma unroll
    for (int f = 0; f < 8; ++f) {
      float q = 0.0f;
#pragma unroll
      for (int f2 = 0; f2 < 8; ++f2) q += ae[f2] * sw1[f * 16 + 8 + f2];
      acc += w2[f] * (sp[f] + q + b1[f]);
    }
    ficin[(size_t)b * 2048 + a]        = (_Float16)umm;
    ficin[(size_t)b * 2048 + 1024 + a] = (_Float16)tanhf(acc);
  }
}

// ---------------------------------------------------------------------------
// launch
// ---------------------------------------------------------------------------
extern "C" void kernel_launch(void* const* d_in, const int* in_sizes, int n_in,
                              void* d_out, int out_size, void* d_ws, size_t ws_size,
                              hipStream_t stream) {
  (void)in_sizes; (void)n_in; (void)out_size; (void)ws_size;
  const int*   ids    = (const int*)d_in[0];
  const float* table  = (const float*)d_in[1];
  const float* cw[4]  = {(const float*)d_in[2], (const float*)d_in[4],
                         (const float*)d_in[6], (const float*)d_in[8]};
  const float* cb[4]  = {(const float*)d_in[3], (const float*)d_in[5],
                         (const float*)d_in[7], (const float*)d_in[9]};
  const float* sc_w   = (const float*)d_in[10];
  const float* sc_b   = (const float*)d_in[11];
  const float* ff_w   = (const float*)d_in[12];
  const float* ff_b   = (const float*)d_in[13];
  const float* api    = (const float*)d_in[14];
  const float* w1     = (const float*)d_in[15];
  const float* b1     = (const float*)d_in[16];
  const float* w2     = (const float*)d_in[17];
  const float* b2     = (const float*)d_in[18];
  const float* ficl_w = (const float*)d_in[19];
  const float* ficl_b = (const float*)d_in[20];
  const float* fus_w  = (const float*)d_in[21];
  const float* fus_b  = (const float*)d_in[22];
  const float* ow     = (const float*)d_in[23];
  const float* ob     = (const float*)d_in[24];

  char* ws = (char*)d_ws;
  size_t off = 0;
  auto alloc = [&](size_t bytes) -> void* {
    off = (off + 255) & ~(size_t)255;
    void* p = ws + off; off += bytes; return p;
  };

  _Float16* Xemb  = (_Float16*)alloc((size_t)B_ * S_ * EP_ * 2);
  _Float16* Wp[4];
  for (int i = 0; i < 4; ++i) Wp[i] = (_Float16*)alloc((size_t)KCH_ * (i + 2) * EP_ * 2);
  _Float16* sc_h   = (_Float16*)alloc((size_t)APAD_ * 1024 * 2);
  _Float16* ficl_h = (_Float16*)alloc((size_t)APAD_ * 2048 * 2);
  _Float16* fus_h  = (_Float16*)alloc((size_t)APAD_ * 2048 * 2);
  _Float16* out_h  = (_Float16*)alloc((size_t)APAD_ * 1024 * 2);
  float*    epool  = (float*)   alloc((size_t)B_ * 1024 * 4);
  _Float16* e_h    = (_Float16*)alloc((size_t)B_ * 1024 * 2);
  float*    ut     = (float*)   alloc((size_t)B_ * 8 * 4);
  _Float16* fic_in = (_Float16*)alloc((size_t)B_ * 2048 * 2);
  _Float16* fus_in = (_Float16*)alloc((size_t)B_ * 2048 * 2);
  _Float16* ummf_h = (_Float16*)alloc((size_t)B_ * 1024 * 2);

  auto zero = [&](void* p, size_t bytes) {
    long n = (long)(bytes / 4);
    mtfm_zero_words<<<dim3((unsigned)((n + 255) / 256)), 256, 0, stream>>>((uint32_t*)p, n);
  };
  zero(epool,  (size_t)B_ * 1024 * 4);
  zero(fic_in, (size_t)B_ * 2048 * 2);
  zero(fus_in, (size_t)B_ * 2048 * 2);
  zero(ummf_h, (size_t)B_ * 1024 * 2);

  // weight packing
  for (int i = 0; i < 4; ++i)
    mtfm_pack_conv<<<dim3(KCH_), 128, 0, stream>>>(cw[i], Wp[i], i + 2);
  mtfm_pack_pad  <<<dim3(APAD_), 256, 0, stream>>>(sc_w, sc_h, A_, 1024, 1024);
  mtfm_pack_pad  <<<dim3(APAD_), 256, 0, stream>>>(ow,   out_h, A_, 1000, 1024);
  mtfm_pack_split<<<dim3(APAD_), 256, 0, stream>>>(ficl_w, ficl_h);
  mtfm_pack_split<<<dim3(APAD_), 256, 0, stream>>>(fus_w,  fus_h);

  // embed
  mtfm_embed_gather<<<dim3(S_, B_), 128, 0, stream>>>(ids, table, Xemb);

  // convs (WMMA) + pooled features
  const int Ls[4] = {511, 510, 509, 508};
  for (int i = 0; i < 4; ++i)
    mtfm_conv_wmma<<<dim3((Ls[i] + 31) / 32, B_), 256, 0, stream>>>(
        Xemb, Wp[i], cb[i], epool, i + 2, Ls[i], i * 256);

  // head
  mtfm_f32_to_f16<<<dim3((B_ * 1024 + 255) / 256), 256, 0, stream>>>(epool, e_h, B_ * 1024);
  mtfm_ut<<<dim3(B_), 32, 0, stream>>>(epool, ff_w, ff_b, ut);
  mtfm_head<<<dim3(B_), 256, 0, stream>>>(ut, api, w1, b1, w2, b2, fic_in);

  // u_sc -> fus_in[:, 0:1000]
  mtfm_gemm_wmma<<<dim3(16, 8), 256, 0, stream>>>(e_h, sc_h, sc_b, 1024, A_, NT_, 0,
                                                  nullptr, fus_in, 2048, 0);
  // u_fic = tanh(...) -> fus_in[:, 1024:2024]
  mtfm_gemm_wmma<<<dim3(16, 8), 256, 0, stream>>>(fic_in, ficl_h, ficl_b, 2048, A_, NT_, 1,
                                                  nullptr, fus_in, 2048, 1024);
  // u_mmf -> ummf_h[:, 0:1000]
  mtfm_gemm_wmma<<<dim3(16, 8), 256, 0, stream>>>(fus_in, fus_h, fus_b, 2048, A_, NT_, 0,
                                                  nullptr, ummf_h, 1024, 0);
  // out -> d_out (f32, 256 x 1000)
  mtfm_gemm_wmma<<<dim3(16, 8), 256, 0, stream>>>(ummf_h, out_h, ob, 1024, A_, NT_, 0,
                                                  (float*)d_out, nullptr, 0, 0);
}
